// RankAugmentedLinearAttention_25323127177259
// MI455X (gfx1250) — matrix operand order repaired and checked by reading, not compiled, not run on hardware
//
#include <hip/hip_runtime.h>
#include <hip/hip_bf16.h>

typedef __attribute__((ext_vector_type(16))) _Float16 v16h;
typedef __attribute__((ext_vector_type(8)))  float    v8f;
typedef __attribute__((ext_vector_type(4)))  int      i32x4;

#define WMMA_F16(a, b, c) \
  __builtin_amdgcn_wmma_f32_16x16x32_f16(false, (a), false, (b), (short)0, (c), false, false)

#if __has_builtin(__builtin_amdgcn_global_load_async_to_lds_b128)
#define RALA_ASYNC_LDS 1
typedef __attribute__((address_space(1))) i32x4 g_i32x4;   // global (prints as __device__)
typedef __attribute__((address_space(3))) i32x4 l_i32x4;   // LDS
#endif

// Problem constants (match reference)
constexpr int BATCH = 4;
constexpr int SEQ   = 4096;
constexpr int EMB   = 512;
constexpr int HEADS = 8;
constexpr int HD    = 64;
constexpr int RTOT  = BATCH * SEQ;   // 16384 token rows

// ---------------- helpers ----------------

__device__ __forceinline__ v8f vzero() {
  v8f z;
#pragma unroll
  for (int i = 0; i < 8; ++i) z[i] = 0.0f;
  return z;
}

__device__ __forceinline__ float elu1(float x) {           // elu(x)+1
  return x > 0.0f ? x + 1.0f : __expf(x);
}
__device__ __forceinline__ float gelu_exact(float x) {     // exact GELU (erf form)
  return 0.5f * x * (1.0f + erff(x * 0.70710678118654752f));
}

// A-operand (16x32 f16): lane = M row (lane&15); halves: lanes0-15 K={0..7,16..23},
// lanes16-31 K={8..15,24..31}  -> element j<8: K=kb+j ; j>=8: K=16+kb+(j-8)
__device__ __forceinline__ v16h load_a_f32(const float* A, int lda, int row, int k0, int lane) {
  const float* p = A + row * lda + k0 + ((lane >> 4) << 3);
  v16h r;
#pragma unroll
  for (int j = 0; j < 8; ++j) { r[j] = (_Float16)p[j]; r[8 + j] = (_Float16)p[16 + j]; }
  return r;
}
__device__ __forceinline__ v16h load_a_f16(const _Float16* A, int lda, int row, int k0, int lane) {
  const _Float16* p = A + row * lda + k0 + ((lane >> 4) << 3);
  v16h r;
#pragma unroll
  for (int j = 0; j < 8; ++j) { r[j] = p[j]; r[8 + j] = p[16 + j]; }
  return r;
}
// B-operand (32x16 f16): lane = N col (lane&15); element j<8: K=kb+j ; j>=8: K=16+kb+(j-8)
__device__ __forceinline__ v16h load_b_f16(const _Float16* Bm, int ldb, int k0, int n0, int lane) {
  const _Float16* p = Bm + (k0 + ((lane >> 4) << 3)) * ldb + n0 + (lane & 15);
  v16h r;
#pragma unroll
  for (int j = 0; j < 8; ++j) { r[j] = p[j * ldb]; r[8 + j] = p[(16 + j) * ldb]; }
  return r;
}

// ---------------- kernel: zero accumulators ----------------

__global__ void rala_zero_kernel(float* p, int n) {
  int i = blockIdx.x * blockDim.x + threadIdx.x;
  if (i < n) p[i] = 0.0f;
}

// ---------------- kernel: f32 -> f16 bulk convert (weights) ----------------

__global__ void rala_cvt_kernel(const float* __restrict__ s, _Float16* __restrict__ d, int n) {
  int idx = (blockIdx.x * blockDim.x + threadIdx.x) * 4;
  if (idx + 3 < n) {
    float4 v = *(const float4*)(s + idx);
    d[idx + 0] = (_Float16)v.x;
    d[idx + 1] = (_Float16)v.y;
    d[idx + 2] = (_Float16)v.z;
    d[idx + 3] = (_Float16)v.w;
  }
}

// ---------------- generic 512-K projection GEMM ----------------
// Y[R,N] = act(X[R,K] @ W[K,N] + bias) ; block = 128x64 tile, 4 waves,
// wave w owns rows [32w,32w+32) (two 16-row tiles) and all 4 column tiles.
// Explicit register double-buffering: next k-chunk A/B loads issue while the
// current 8 WMMAs execute, so waits are partial instead of loadcnt==0 stalls.
template <int ACT, bool AF16, bool OUTF32>
__global__ __launch_bounds__(128) void rala_gemm_kernel(
    const void* __restrict__ Ain, const _Float16* __restrict__ Wh,
    const float* __restrict__ bias, void* __restrict__ Out,
    int R, int K, int N) {
  const int lane = threadIdx.x & 31;
  const int wave = threadIdx.x >> 5;
  const int r0   = blockIdx.x * 128 + wave * 32;
  const int n0   = blockIdx.y * 64;

  v8f acc[2][4];
#pragma unroll
  for (int rt = 0; rt < 2; ++rt)
#pragma unroll
    for (int c = 0; c < 4; ++c) acc[rt][c] = vzero();

  const int arow = r0 + (lane & 15);

  v16h a0, a1, b[4];
  if constexpr (AF16) {
    a0 = load_a_f16((const _Float16*)Ain, K, arow,      0, lane);
    a1 = load_a_f16((const _Float16*)Ain, K, arow + 16, 0, lane);
  } else {
    a0 = load_a_f32((const float*)Ain, K, arow,      0, lane);
    a1 = load_a_f32((const float*)Ain, K, arow + 16, 0, lane);
  }
#pragma unroll
  for (int c = 0; c < 4; ++c) b[c] = load_b_f16(Wh, N, 0, n0 + c * 16, lane);

  for (int k = 0; k < K - 32; k += 32) {
    const int kn = k + 32;
    if (kn + 32 < K) __builtin_prefetch(Wh + (kn + 32 + lane) * N + n0, 0, 0);
    v16h na0, na1, nb[4];
    if constexpr (AF16) {
      na0 = load_a_f16((const _Float16*)Ain, K, arow,      kn, lane);
      na1 = load_a_f16((const _Float16*)Ain, K, arow + 16, kn, lane);
    } else {
      na0 = load_a_f32((const float*)Ain, K, arow,      kn, lane);
      na1 = load_a_f32((const float*)Ain, K, arow + 16, kn, lane);
    }
#pragma unroll
    for (int c = 0; c < 4; ++c) nb[c] = load_b_f16(Wh, N, kn, n0 + c * 16, lane);
#pragma unroll
    for (int c = 0; c < 4; ++c) {
      acc[0][c] = WMMA_F16(a0, b[c], acc[0][c]);
      acc[1][c] = WMMA_F16(a1, b[c], acc[1][c]);
    }
    a0 = na0; a1 = na1;
#pragma unroll
    for (int c = 0; c < 4; ++c) b[c] = nb[c];
  }
#pragma unroll
  for (int c = 0; c < 4; ++c) {  // final chunk
    acc[0][c] = WMMA_F16(a0, b[c], acc[0][c]);
    acc[1][c] = WMMA_F16(a1, b[c], acc[1][c]);
  }

#pragma unroll
  for (int rt = 0; rt < 2; ++rt) {
#pragma unroll
    for (int c = 0; c < 4; ++c) {
      const int col = n0 + c * 16 + (lane & 15);
      const float bb = bias[col];
#pragma unroll
      for (int i = 0; i < 8; ++i) {
        const int row = r0 + rt * 16 + i + ((lane >> 4) << 3);
        float v = acc[rt][c][i] + bb;
        if (ACT == 1) v = elu1(v);
        if constexpr (OUTF32) ((float*)Out)[row * N + col] = v;
        else                  ((_Float16*)Out)[row * N + col] = (_Float16)v;
      }
    }
  }
}

// ---------------- per-head key MLP: elu1(gelu(k@W1+b1)@W2+b2) ----------------
// block: 64 token rows x one head; stage1 K=64 -> hidden 128 (LDS), stage2 K=128 -> 64
__global__ __launch_bounds__(128) void rala_kmlp_kernel(
    const _Float16* __restrict__ kf, const _Float16* __restrict__ W1h,
    const float* __restrict__ b1, const _Float16* __restrict__ W2h,
    const float* __restrict__ b2, _Float16* __restrict__ kaug) {
  __shared__ _Float16 hid[64][136];  // 128 + pad 8 to break bank alignment
  const int lane = threadIdx.x & 31;
  const int wave = threadIdx.x >> 5;
  const int r0   = blockIdx.x * 64;
  const int h    = blockIdx.y;
  const int ch   = h * HD;
  const int rw   = r0 + wave * 16;

  // stage 1: hid = gelu(k @ W1 + b1)
  v8f acc1[8];
#pragma unroll
  for (int c = 0; c < 8; ++c) acc1[c] = vzero();
  for (int k = 0; k < 64; k += 32) {
    v16h a = load_a_f16(kf, EMB, rw + (lane & 15), ch + k, lane);
#pragma unroll
    for (int c = 0; c < 8; ++c) {
      v16h b = load_b_f16(W1h, 128, k, c * 16, lane);
      acc1[c] = WMMA_F16(a, b, acc1[c]);
    }
  }
#pragma unroll
  for (int c = 0; c < 8; ++c) {
    const int col = c * 16 + (lane & 15);
    const float bb = b1[col];
#pragma unroll
    for (int i = 0; i < 8; ++i) {
      const int row = wave * 16 + i + ((lane >> 4) << 3);
      hid[row][col] = (_Float16)gelu_exact(acc1[c][i] + bb);
    }
  }
  __syncthreads();

  // stage 2: kaug = elu1(hid @ W2 + b2)
  v8f acc2[4];
#pragma unroll
  for (int c = 0; c < 4; ++c) acc2[c] = vzero();
  for (int k = 0; k < 128; k += 32) {
    v16h a = load_a_f16(&hid[0][0], 136, wave * 16 + (lane & 15), k, lane);
#pragma unroll
    for (int c = 0; c < 4; ++c) {
      v16h b = load_b_f16(W2h, 64, k, c * 16, lane);
      acc2[c] = WMMA_F16(a, b, acc2[c]);
    }
  }
#pragma unroll
  for (int c = 0; c < 4; ++c) {
    const int col = c * 16 + (lane & 15);
    const float bb = b2[col];
#pragma unroll
    for (int i = 0; i < 8; ++i) {
      const int row = rw + i + ((lane >> 4) << 3);
      kaug[row * EMB + ch + col] = (_Float16)elu1(acc2[c][i] + bb);
    }
  }
}

// ---------------- kv_context = K'^T V / M  and  ksum = sum_m K' / M ----------------
// grid (msplit=16, B, H); each block reduces 256 rows of m, atomically adds partials.
// Staging uses CDNA5 async global->LDS copies (ASYNCcnt) when available.
__global__ __launch_bounds__(128) void rala_kvctx_kernel(
    const _Float16* __restrict__ kaug, const _Float16* __restrict__ vf,
    float* __restrict__ kv, float* __restrict__ ksum) {
  __shared__ _Float16 Kt[32][72];   // row stride 144B (16B-aligned), pad for transposed reads
  __shared__ _Float16 Vt[32][64];
  const int lane = threadIdx.x & 31;
  const int wave = threadIdx.x >> 5;
  const int t    = threadIdx.x;
  const int b = blockIdx.y, h = blockIdx.z;
  const int m0 = blockIdx.x * 256;
  const int ch = h * HD;
  const int rowbase = b * SEQ;

  v8f acc[4];
#pragma unroll
  for (int c = 0; c < 4; ++c) acc[c] = vzero();
  float ks = 0.0f;

  for (int mc = 0; mc < 256; mc += 32) {
    __syncthreads();
    {  // cooperative stage: 32 rows x 64 cols of K' and V (f16)
      const int row = t >> 2;
      const int col = (t & 3) * 16;
      const int gr = rowbase + m0 + mc + row;
      const _Float16* kp = kaug + gr * EMB + ch + col;
      const _Float16* vp = vf + gr * EMB + ch + col;
#ifdef RALA_ASYNC_LDS
      __builtin_amdgcn_global_load_async_to_lds_b128(
          (g_i32x4*)kp, (l_i32x4*)&Kt[row][col], 0, 0);
      __builtin_amdgcn_global_load_async_to_lds_b128(
          (g_i32x4*)(kp + 8), (l_i32x4*)&Kt[row][col + 8], 0, 0);
      __builtin_amdgcn_global_load_async_to_lds_b128(
          (g_i32x4*)vp, (l_i32x4*)&Vt[row][col], 0, 0);
      __builtin_amdgcn_global_load_async_to_lds_b128(
          (g_i32x4*)(vp + 8), (l_i32x4*)&Vt[row][col + 8], 0, 0);
#if __has_builtin(__builtin_amdgcn_s_wait_asynccnt)
      __builtin_amdgcn_s_wait_asynccnt(0);
#else
      asm volatile("s_wait_asynccnt 0x0" ::: "memory");
#endif
#else
#pragma unroll
      for (int j = 0; j < 16; ++j) { Kt[row][col + j] = kp[j]; Vt[row][col + j] = vp[j]; }
#endif
    }
    __syncthreads();

    // A = K'^T tile: row index = d (lane&15 within dstrip), K index = m (transposed LDS read)
    v16h a;
    {
      const int d = wave * 16 + (lane & 15);
      const int kb = (lane >> 4) << 3;
#pragma unroll
      for (int j = 0; j < 8; ++j) { a[j] = Kt[kb + j][d]; a[8 + j] = Kt[16 + kb + j][d]; }
    }
#pragma unroll
    for (int c = 0; c < 4; ++c) {
      v16h bm;  // B = V: lane = e col (lane&15), elements over m in table order
      const int kbv = (lane >> 4) << 3;
#pragma unroll
      for (int j = 0; j < 8; ++j) { bm[j] = Vt[kbv + j][c * 16 + (lane & 15)]; bm[8 + j] = Vt[16 + kbv + j][c * 16 + (lane & 15)]; }
      acc[c] = WMMA_F16(a, bm, acc[c]);
    }
    if (t < 64) {  // column sums of K' for the denominator
#pragma unroll
      for (int mm = 0; mm < 32; ++mm) ks += (float)Kt[mm][t];
    }
  }

  const float sc = 1.0f / (float)SEQ;
  const int bh = b * HEADS + h;
#pragma unroll
  for (int c = 0; c < 4; ++c) {
    const int e = c * 16 + (lane & 15);
#pragma unroll
    for (int i = 0; i < 8; ++i) {
      const int d = wave * 16 + i + ((lane >> 4) << 3);
      atomicAdd(&kv[(bh * 64 + d) * 64 + e], acc[c][i] * sc);
    }
  }
  if (t < 64) atomicAdd(&ksum[bh * 64 + t], ks * sc);
}

// ---------------- out = (Q @ kv) / clip(Q . ksum, 1e-6), nan/clamp, -> f16 ----------------
__global__ __launch_bounds__(128) void rala_attn_kernel(
    const _Float16* __restrict__ qf, const float* __restrict__ kv,
    const float* __restrict__ ksum, _Float16* __restrict__ attn) {
  __shared__ float den[64];
  const int lane = threadIdx.x & 31;
  const int wave = threadIdx.x >> 5;
  const int t    = threadIdx.x;
  const int r0   = blockIdx.x * 64;
  const int h    = blockIdx.y;
  const int b    = r0 / SEQ;           // 64 | SEQ, so uniform per block
  const int ch   = h * HD;
  const int bh   = b * HEADS + h;

  if (t < 64) {
    const _Float16* qp = qf + (r0 + t) * EMB + ch;
    const float* kp = ksum + bh * 64;
    float s = 0.0f;
#pragma unroll
    for (int d = 0; d < 64; ++d) s += (float)qp[d] * kp[d];
    den[t] = fmaxf(s, 1e-6f);
  }
  __syncthreads();

  const int rw = r0 + wave * 16;
  v8f acc[4];
#pragma unroll
  for (int c = 0; c < 4; ++c) acc[c] = vzero();
  for (int k = 0; k < 64; k += 32) {
    v16h a = load_a_f16(qf, EMB, rw + (lane & 15), ch + k, lane);
#pragma unroll
    for (int c = 0; c < 4; ++c) {
      const float* bp = kv + (bh * 64 + k + ((lane >> 4) << 3)) * 64 + c * 16 + (lane & 15);
      v16h bm;
#pragma unroll
      for (int j = 0; j < 8; ++j) { bm[j] = (_Float16)bp[j * 64]; bm[8 + j] = (_Float16)bp[(16 + j) * 64]; }
      acc[c] = WMMA_F16(a, bm, acc[c]);
    }
  }
#pragma unroll
  for (int c = 0; c < 4; ++c) {
    const int col = c * 16 + (lane & 15);
#pragma unroll
    for (int i = 0; i < 8; ++i) {
      const int lr = wave * 16 + i + ((lane >> 4) << 3);
      float v = acc[c][i] / den[lr];
      v = (v != v) ? 0.0f : fminf(fmaxf(v, -65000.0f), 65000.0f);
      attn[(r0 + lr) * EMB + ch + col] = (_Float16)v;
    }
  }
}

// ---------------- host launch ----------------

extern "C" void kernel_launch(void* const* d_in, const int* in_sizes, int n_in,
                              void* d_out, int out_size, void* d_ws, size_t ws_size,
                              hipStream_t stream) {
  (void)in_sizes; (void)n_in; (void)out_size; (void)ws_size;
  const float* query = (const float*)d_in[0];
  const float* keyi  = (const float*)d_in[1];
  const float* value = (const float*)d_in[2];
  const float* Wq = (const float*)d_in[3];  const float* bq = (const float*)d_in[4];
  const float* Wk = (const float*)d_in[5];  const float* bk = (const float*)d_in[6];
  const float* Wv = (const float*)d_in[7];  const float* bv = (const float*)d_in[8];
  const float* W1 = (const float*)d_in[9];  const float* b1 = (const float*)d_in[10];
  const float* W2 = (const float*)d_in[11]; const float* b2 = (const float*)d_in[12];
  const float* Wo = (const float*)d_in[13]; const float* bo = (const float*)d_in[14];

  // workspace layout (bytes):
  //   5 x f16 [16384,512] activation buffers, f32 kv+ksum, then f16 weights
  const size_t SZ16 = (size_t)RTOT * EMB * sizeof(_Float16);  // 16 MB
  char* ws = (char*)d_ws;
  _Float16* qf   = (_Float16*)(ws + 0 * SZ16);
  _Float16* kf   = (_Float16*)(ws + 1 * SZ16);
  _Float16* vf   = (_Float16*)(ws + 2 * SZ16);
  _Float16* kaug = (_Float16*)(ws + 3 * SZ16);
  _Float16* attn = (_Float16*)(ws + 4 * SZ16);
  float* kv   = (float*)(ws + 5 * SZ16);                      // [B,H,64,64]
  float* ksum = kv + (size_t)BATCH * HEADS * 64 * 64;         // [B,H,64]
  char* wbase = ws + 5 * SZ16 + (1u << 20);                   // 1MB pad past kv/ksum
  const size_t WSZ = (size_t)EMB * EMB * sizeof(_Float16);    // 512 KB
  _Float16* Wqh = (_Float16*)(wbase + 0 * WSZ);
  _Float16* Wkh = (_Float16*)(wbase + 1 * WSZ);
  _Float16* Wvh = (_Float16*)(wbase + 2 * WSZ);
  _Float16* Woh = (_Float16*)(wbase + 3 * WSZ);
  _Float16* W1h = (_Float16*)(wbase + 4 * WSZ);
  _Float16* W2h = W1h + HD * 2 * HD;

  // zero kv/ksum accumulators (contiguous)
  const int nz = BATCH * HEADS * 64 * 64 + BATCH * HEADS * 64;
  rala_zero_kernel<<<(nz + 255) / 256, 256, 0, stream>>>(kv, nz);

  // pre-convert weights to f16 (keeps cvt out of GEMM hot loops)
  const int NW = EMB * EMB;
  rala_cvt_kernel<<<(NW / 4 + 255) / 256, 256, 0, stream>>>(Wq, Wqh, NW);
  rala_cvt_kernel<<<(NW / 4 + 255) / 256, 256, 0, stream>>>(Wk, Wkh, NW);
  rala_cvt_kernel<<<(NW / 4 + 255) / 256, 256, 0, stream>>>(Wv, Wvh, NW);
  rala_cvt_kernel<<<(NW / 4 + 255) / 256, 256, 0, stream>>>(Wo, Woh, NW);
  const int NW1 = HD * HD * 2;
  rala_cvt_kernel<<<(NW1 / 4 + 255) / 256, 256, 0, stream>>>(W1, W1h, NW1);
  rala_cvt_kernel<<<(NW1 / 4 + 255) / 256, 256, 0, stream>>>(W2, W2h, NW1);

  dim3 gproj(RTOT / 128, EMB / 64);
  rala_gemm_kernel<1, false, false><<<gproj, 128, 0, stream>>>(query, Wqh, bq, qf, RTOT, EMB, EMB);
  rala_gemm_kernel<1, false, false><<<gproj, 128, 0, stream>>>(keyi,  Wkh, bk, kf, RTOT, EMB, EMB);
  rala_gemm_kernel<0, false, false><<<gproj, 128, 0, stream>>>(value, Wvh, bv, vf, RTOT, EMB, EMB);

  rala_kmlp_kernel<<<dim3(RTOT / 64, HEADS), 128, 0, stream>>>(kf, W1h, b1, W2h, b2, kaug);
  rala_kvctx_kernel<<<dim3(16, BATCH, HEADS), 128, 0, stream>>>(kaug, vf, kv, ksum);
  rala_attn_kernel<<<dim3(RTOT / 64, HEADS), 128, 0, stream>>>(qf, kv, ksum, attn);

  rala_gemm_kernel<0, true, true><<<gproj, 128, 0, stream>>>(attn, Woh, bo, d_out, RTOT, EMB, EMB);
}
